// GumbelLatentVQ_72576357368196
// MI455X (gfx1250) — compile-verified, matrix operand order checked by reading, not run on hardware
//
#include <hip/hip_runtime.h>
#include <hip/hip_bf16.h>
#include <math.h>

// ---------------------------------------------------------------------------
// GumbelLatentVQ forward for MI455X (gfx1250).
// All GEMMs run on v_wmma_f32_16x16x32_bf16 (wave32, 16x16 tiles, f32 acc).
// VQ softmax is a branchless three-pass sweep (max/argmax, sumexp, probs).
// ---------------------------------------------------------------------------

typedef __bf16 bf16_t;
typedef __attribute__((ext_vector_type(16))) __bf16 v16bf;
typedef __attribute__((ext_vector_type(8)))  float  v8f;

#define WMMA_BF16(a,b,c) \
  __builtin_amdgcn_wmma_f32_16x16x32_bf16(false,(a),false,(b),(short)0,(c),false,false)

static constexpr int    BB   = 32;
static constexpr int    HH   = 32;
static constexpr int    WW   = 32;
static constexpr int    DIN  = 512;
static constexpr int    NE   = 8192;
static constexpr int    NTOK = BB * HH * WW;   // 32768 encoder tokens
static constexpr int    NQ   = BB * 16 * 16;   // 8192 VQ tokens
static constexpr size_t OUT_DEC = (size_t)BB * HH * WW * DIN;  // 16,777,216

// ---- workspace layout (bytes), with region reuse ---------------------------
static constexpr size_t OFF_DELTA = 0;                               // bf16 [NTOK,512]
static constexpr size_t OFF_X1    = 0;                               // bf16 [NTOK,512] (reuse: delta dead)
static constexpr size_t OFF_Y1    = (size_t)NTOK * 512 * 2;          // f32  [NTOK,512]
static constexpr size_t OFF_X2    = OFF_Y1;                          // bf16 [NTOK,128] (reuse: y1 dead)
static constexpr size_t OFF_Y3    = OFF_X2 + (size_t)NTOK * 128 * 2; // f32  [NTOK,64]
static constexpr size_t OFF_X3    = OFF_Y3 + (size_t)NTOK * 64 * 4;  // f32  [NTOK,64]
static constexpr size_t OFF_POOLF = OFF_X3 + (size_t)NTOK * 64 * 4;  // f32  [NQ,64]
static constexpr size_t OFF_POOLB = OFF_POOLF + (size_t)NQ * 64 * 4; // bf16 [NQ,64]
static constexpr size_t OFF_QB    = OFF_POOLB + (size_t)NQ * 64 * 2; // bf16 [NQ,64]
static constexpr size_t OFF_H1    = OFF_QB    + (size_t)NQ * 64 * 2; // bf16 [NQ,128]
static constexpr size_t OFF_YDEC  = OFF_H1    + (size_t)NQ * 128 * 2;// f32  [NQ,512]
// persistent weights after the y1 region (y1 spans [OFF_Y1, OFF_Y1+NTOK*512*4))
static constexpr size_t OFF_W1T   = OFF_Y1 + (size_t)NTOK * 512 * 4; // bf16 [512,512]
static constexpr size_t OFF_W2T   = OFF_W1T  + (size_t)512 * 512 * 2;// bf16 [128,512]
static constexpr size_t OFF_W3T   = OFF_W2T  + (size_t)128 * 512 * 2;// bf16 [64,128]
static constexpr size_t OFF_DW1T  = OFF_W3T  + (size_t)64  * 128 * 2;// bf16 [128,64]
static constexpr size_t OFF_DW2T  = OFF_DW1T + (size_t)128 * 64  * 2;// bf16 [512,128]
static constexpr size_t OFF_EMBB  = OFF_DW2T + (size_t)512 * 128 * 2;// bf16 [8192,64]
static constexpr size_t OFF_EHALF = OFF_EMBB + (size_t)NE * 64 * 2;  // f32  [8192] = 0.5*|e|^2
static constexpr size_t OFF_AVG   = OFF_EHALF+ (size_t)NE * 4;       // f32  [8192] prob accum
static constexpr size_t OFF_CMT   = OFF_AVG  + (size_t)NE * 4;       // f32  [1]    commit accum
static constexpr size_t OFF_IDX   = OFF_CMT  + 64;                   // i32  [8192] argmax indices

// ---------------------------------------------------------------------------
__device__ __forceinline__ v16bf load_frag32(const bf16_t* p) {
  union { v16bf v; uint4 u[2]; } t;
  t.u[0] = *(const uint4*)p;
  t.u[1] = *(const uint4*)(p + 16);
  return t.v;
}

__device__ __forceinline__ float wave_sum(float v) {
#pragma unroll
  for (int o = 16; o > 0; o >>= 1) v += __shfl_xor(v, o, 32);
  return v;
}

__device__ __forceinline__ float gelu_exact(float x) {
  return 0.5f * x * (1.0f + erff(x * 0.70710678118654752f));
}

// ---------------------------------------------------------------------------
// Generic 1-wave WMMA GEMM: C[M,N] = A[M,K](bf16) * BT[N,K]^T(bf16) (+bias,GELU)
// Each block (32 threads) computes a 16 x (16*NSUB) tile.
// ---------------------------------------------------------------------------
template<int NSUB, bool BIAS, bool ACT, bool OUTB>
__global__ __launch_bounds__(32)
void gemm_wmma(const bf16_t* __restrict__ A, const bf16_t* __restrict__ BT,
               const float* __restrict__ bias,
               float* __restrict__ Cf, bf16_t* __restrict__ Cb,
               int N, int K) {
  int lane = threadIdx.x;
  int r    = lane & 15;
  int koff = (lane >> 4) * 8;
  size_t m0 = (size_t)blockIdx.x * 16;
  int    n0 = blockIdx.y * (16 * NSUB);
  const bf16_t* Ab = A + (m0 + r) * (size_t)K + koff;

  v8f acc[NSUB] = {};
  for (int k0 = 0; k0 < K; k0 += 32) {
    v16bf a = load_frag32(Ab + k0);
#pragma unroll
    for (int i = 0; i < NSUB; i++) {
      const bf16_t* Bb = BT + (size_t)(n0 + i * 16 + r) * K + koff + k0;
      v16bf b = load_frag32(Bb);
      acc[i] = WMMA_BF16(a, b, acc[i]);
    }
  }
  size_t mrow = m0 + (size_t)(lane >> 4) * 8;
#pragma unroll
  for (int i = 0; i < NSUB; i++) {
    int n = n0 + i * 16 + r;
    float bv = BIAS ? bias[n] : 0.0f;
#pragma unroll
    for (int k = 0; k < 8; k++) {
      float v = acc[i][k] + bv;
      if (ACT) v = gelu_exact(v);
      size_t o = (mrow + k) * (size_t)N + n;
      if (OUTB) Cb[o] = (bf16_t)v; else Cf[o] = v;
    }
  }
}

// ---------------------------------------------------------------------------
// Prep kernels
// ---------------------------------------------------------------------------
__global__ void delta_bf16_k(const float* __restrict__ t0, const float* __restrict__ t1,
                             bf16_t* __restrict__ o, size_t n) {
  size_t i = (size_t)blockIdx.x * 256 + threadIdx.x;
  if (i < n) o[i] = (bf16_t)(t1[i] - t0[i]);
}

// out[n*K+k] = W[k*N+n]  (W is [K,N] row-major; out is N x K "B-transposed")
__global__ void transp_bf16_k(const float* __restrict__ W, bf16_t* __restrict__ o,
                              int K, int N) {
  int i = blockIdx.x * 256 + threadIdx.x;
  if (i < K * N) {
    int n = i / K, k = i - n * K;
    o[i] = (bf16_t)W[(size_t)k * N + n];
  }
}

__global__ __launch_bounds__(32)
void emb_prep_k(const float* __restrict__ emb, bf16_t* __restrict__ eb,
                float* __restrict__ eh) {
  int row = blockIdx.x, l = threadIdx.x;
  const float* p = emb + (size_t)row * 64;
  float v0 = p[l], v1 = p[l + 32];
  bf16_t* o = eb + (size_t)row * 64;
  o[l] = (bf16_t)v0; o[l + 32] = (bf16_t)v1;
  float ns = wave_sum(v0 * v0 + v1 * v1);
  if (l == 0) eh[row] = 0.5f * ns;
}

__global__ void zero_k(float* p, int n) {
  int i = blockIdx.x * 256 + threadIdx.x;
  if (i < n) p[i] = 0.0f;
}

// ---------------------------------------------------------------------------
// LayerNorm kernels
// ---------------------------------------------------------------------------
__global__ __launch_bounds__(128)
void ln1_gelu_k(const float* __restrict__ y, const float* __restrict__ g,
                const float* __restrict__ b, bf16_t* __restrict__ out) {
  int row = blockIdx.x, t = threadIdx.x;
  const float* yr = y + (size_t)row * 512;
  float v[4]; float s = 0.f, s2 = 0.f;
#pragma unroll
  for (int i = 0; i < 4; i++) { v[i] = yr[t + 128 * i]; s += v[i]; s2 += v[i] * v[i]; }
  __shared__ float rs[4], rs2[4];
  s = wave_sum(s); s2 = wave_sum(s2);
  int w = t >> 5;
  if ((t & 31) == 0) { rs[w] = s; rs2[w] = s2; }
  __syncthreads();
  s  = rs[0] + rs[1] + rs[2] + rs[3];
  s2 = rs2[0] + rs2[1] + rs2[2] + rs2[3];
  float mu   = s * (1.0f / 512.0f);
  float var  = s2 * (1.0f / 512.0f) - mu * mu;
  float rstd = rsqrtf(var + 1e-5f);
  bf16_t* orow = out + (size_t)row * 512;
#pragma unroll
  for (int i = 0; i < 4; i++) {
    int c = t + 128 * i;
    float xv = (v[i] - mu) * rstd * g[c] + b[c];
    orow[c] = (bf16_t)gelu_exact(xv);
  }
}

__global__ __launch_bounds__(32)
void ln2_k(const float* __restrict__ y, const float* __restrict__ g,
           const float* __restrict__ b, float* __restrict__ out) {
  int row = blockIdx.x, l = threadIdx.x;
  const float* yr = y + (size_t)row * 64;
  float v0 = yr[l], v1 = yr[l + 32];
  float s  = wave_sum(v0 + v1);
  float s2 = wave_sum(v0 * v0 + v1 * v1);
  float mu   = s * (1.0f / 64.0f);
  float var  = s2 * (1.0f / 64.0f) - mu * mu;
  float rstd = rsqrtf(var + 1e-5f);
  float* orow = out + (size_t)row * 64;
  orow[l]      = (v0 - mu) * rstd * g[l] + b[l];
  orow[l + 32] = (v1 - mu) * rstd * g[l + 32] + b[l + 32];
}

// ---------------------------------------------------------------------------
// 2x2 avg pool + L2-normalize-to-sqrt(64)
// ---------------------------------------------------------------------------
__global__ __launch_bounds__(32)
void pool_norm_k(const float* __restrict__ x3, float* __restrict__ pf,
                 bf16_t* __restrict__ pb) {
  int t = blockIdx.x, l = threadIdx.x;          // token 0..8191
  int gx = t & 15, gy = (t >> 4) & 15, bb = t >> 8;
  float a0 = 0.f, a1 = 0.f;
#pragma unroll
  for (int dy = 0; dy < 2; dy++)
#pragma unroll
    for (int dx = 0; dx < 2; dx++) {
      const float* p = x3 + ((size_t)((bb * 32 + gy * 2 + dy) * 32) + gx * 2 + dx) * 64;
      a0 += p[l]; a1 += p[l + 32];
    }
  a0 *= 0.25f; a1 *= 0.25f;
  float ns  = wave_sum(a0 * a0 + a1 * a1);
  float sc  = 8.0f / fmaxf(sqrtf(ns), 1e-12f);
  a0 *= sc; a1 *= sc;
  float*  of = pf + (size_t)t * 64;
  bf16_t* ob = pb + (size_t)t * 64;
  of[l] = a0; of[l + 32] = a1;
  ob[l] = (bf16_t)a0; ob[l + 32] = (bf16_t)a1;
}

// ---------------------------------------------------------------------------
// VQ: per-wave 16 tokens vs all 8192 codes via WMMA.
// logits = x.e - 0.5|e|^2 (row-constant -32 and /T fold out; T=2 via -dist/2).
// Branchless three-pass: (1) max+argmax, (2) sumexp, (3) prob accumulation.
// ---------------------------------------------------------------------------
__global__ __launch_bounds__(32)
void vq_k(const bf16_t* __restrict__ xq, const bf16_t* __restrict__ eb,
          const float* __restrict__ ehalf, const float* __restrict__ poolf,
          const float* __restrict__ embf, float* __restrict__ avg_acc,
          float* __restrict__ cmt_acc, int* __restrict__ idx_i32,
          float* __restrict__ idx_f32, bf16_t* __restrict__ qb) {
  __shared__ float lacc[8192];
  int lane = threadIdx.x;
  int tok0 = blockIdx.x * 16;
  for (int i = lane; i < 8192; i += 32) lacc[i] = 0.0f;
  __syncthreads();

  int r = lane & 15, koff = (lane >> 4) * 8;
  const bf16_t* Ab = xq + (size_t)(tok0 + r) * 64 + koff;
  v16bf a0 = load_frag32(Ab);
  v16bf a1 = load_frag32(Ab + 32);

  float m[8]; int am[8];
#pragma unroll
  for (int k = 0; k < 8; k++) { m[k] = -3.4e38f; am[k] = 0; }

  // ---- pass 1: row max + argmax (branchless: fmax + cndmask) ----
  for (int j = 0; j < 512; j++) {
    const bf16_t* Bb = eb + (size_t)(j * 16 + r) * 64 + koff;
    v16bf b0 = load_frag32(Bb);
    v16bf b1 = load_frag32(Bb + 32);
    v8f acc = {};
    acc = WMMA_BF16(a0, b0, acc);
    acc = WMMA_BF16(a1, b1, acc);
    int col = j * 16 + r;
    float h = ehalf[col];
#pragma unroll
    for (int k = 0; k < 8; k++) {
      float lg = acc[k] - h;
      am[k] = (lg > m[k]) ? col : am[k];   // strict > keeps first occurrence
      m[k]  = fmaxf(m[k], lg);
    }
  }
  // merge (max, argmax) across the 16 lanes of each half (tie -> lower index)
#pragma unroll
  for (int o = 1; o < 16; o <<= 1) {
#pragma unroll
    for (int k = 0; k < 8; k++) {
      float m2 = __shfl_xor(m[k], o, 32);
      int   a2 = __shfl_xor(am[k], o, 32);
      bool  take = (m2 > m[k]) || (m2 == m[k] && a2 < am[k]);
      am[k] = take ? a2 : am[k];
      m[k]  = fmaxf(m[k], m2);
    }
  }

  // ---- pass 2: row sum of exp(logit - max) ----
  float s[8];
#pragma unroll
  for (int k = 0; k < 8; k++) s[k] = 0.0f;
  for (int j = 0; j < 512; j++) {
    const bf16_t* Bb = eb + (size_t)(j * 16 + r) * 64 + koff;
    v16bf b0 = load_frag32(Bb);
    v16bf b1 = load_frag32(Bb + 32);
    v8f acc = {};
    acc = WMMA_BF16(a0, b0, acc);
    acc = WMMA_BF16(a1, b1, acc);
    float h = ehalf[j * 16 + r];
#pragma unroll
    for (int k = 0; k < 8; k++) s[k] += __expf(acc[k] - h - m[k]);
  }
#pragma unroll
  for (int o = 1; o < 16; o <<= 1) {
#pragma unroll
    for (int k = 0; k < 8; k++) s[k] += __shfl_xor(s[k], o, 32);
  }
  float rs[8];
#pragma unroll
  for (int k = 0; k < 8; k++) rs[k] = 1.0f / s[k];

  // ---- pass 3: accumulate softmax probs into LDS histogram ----
  for (int j = 0; j < 512; j++) {
    const bf16_t* Bb = eb + (size_t)(j * 16 + r) * 64 + koff;
    v16bf b0 = load_frag32(Bb);
    v16bf b1 = load_frag32(Bb + 32);
    v8f acc = {};
    acc = WMMA_BF16(a0, b0, acc);
    acc = WMMA_BF16(a1, b1, acc);
    int col = j * 16 + r;
    float h = ehalf[col];
    float p = 0.0f;
#pragma unroll
    for (int k = 0; k < 8; k++) p += __expf(acc[k] - h - m[k]) * rs[k];
    atomicAdd(&lacc[col], p);
  }
  __syncthreads();
  for (int i = lane; i < 8192; i += 32) atomicAdd(&avg_acc[i], lacc[i]);

  // ---- indices, codebook gather, commitment partial ----
  float cpart = 0.0f;
#pragma unroll
  for (int rr = 0; rr < 16; rr++) {
    int src = (rr >= 8) ? 16 : 0;
    int idx = __shfl(am[rr & 7], src, 32);
    int tok = tok0 + rr;
    if (lane == 0) { idx_i32[tok] = idx; idx_f32[tok] = (float)idx; }
    const float* e = embf + (size_t)idx * 64;
    const float* f = poolf + (size_t)tok * 64;
    float e0 = e[lane], e1 = e[lane + 32];
    float d0 = e0 - f[lane], d1 = e1 - f[lane + 32];
    cpart += d0 * d0 + d1 * d1;
    bf16_t* q = qb + (size_t)tok * 64;
    q[lane] = (bf16_t)e0; q[lane + 32] = (bf16_t)e1;
  }
  cpart = wave_sum(cpart);
  if (lane == 0) atomicAdd(cmt_acc, cpart);
}

__global__ __launch_bounds__(256)
void vq_finalize_k(const float* __restrict__ avg, const float* __restrict__ cmt,
                   float* __restrict__ out) {
  __shared__ float red[8];
  float ent = 0.0f;
  for (int i = threadIdx.x; i < 8192; i += 256) {
    float p = avg[i] * (1.0f / 8192.0f);
    ent += p * logf(p + 1e-10f);
  }
  ent = wave_sum(ent);
  int w = threadIdx.x >> 5;
  if ((threadIdx.x & 31) == 0) red[w] = ent;
  __syncthreads();
  if (threadIdx.x == 0) {
    float t = 0.f;
    for (int i = 0; i < 8; i++) t += red[i];
    out[OUT_DEC]     = __expf(-t);                          // perplexity
    out[OUT_DEC + 1] = cmt[0] * (1.0f / (8192.0f * 64.0f)); // commitment loss
  }
}

// ---------------------------------------------------------------------------
// 2x bilinear upsample (align_corners=False), channels-last, float4 over C
// ---------------------------------------------------------------------------
__global__ __launch_bounds__(256)
void upsample2x_k(const float* __restrict__ src, float* __restrict__ dst) {
  size_t gid = (size_t)blockIdx.x * 256 + threadIdx.x;
  if (gid >= (size_t)32 * 32 * 32 * 128) return;
  int c4 = gid & 127;
  size_t t = gid >> 7;
  int x = t & 31; t >>= 5;
  int y = t & 31; t >>= 5;
  int bb = (int)t;
  float syf = 0.5f * y - 0.25f;
  float sxf = 0.5f * x - 0.25f;
  int y0 = (int)floorf(syf), x0 = (int)floorf(sxf);
  float wy1 = syf - y0, wx1 = sxf - x0;
  int y0c = max(y0, 0), y1c = min(y0 + 1, 15);
  int x0c = max(x0, 0), x1c = min(x0 + 1, 15);
  const float4* s = (const float4*)src;
  size_t base = (size_t)bb * 16 * 16 * 128;
  float4 v00 = s[base + ((size_t)y0c * 16 + x0c) * 128 + c4];
  float4 v01 = s[base + ((size_t)y0c * 16 + x1c) * 128 + c4];
  float4 v10 = s[base + ((size_t)y1c * 16 + x0c) * 128 + c4];
  float4 v11 = s[base + ((size_t)y1c * 16 + x1c) * 128 + c4];
  float w00 = (1.f - wy1) * (1.f - wx1), w01 = (1.f - wy1) * wx1;
  float w10 = wy1 * (1.f - wx1),         w11 = wy1 * wx1;
  float4 o;
  o.x = w00 * v00.x + w01 * v01.x + w10 * v10.x + w11 * v11.x;
  o.y = w00 * v00.y + w01 * v01.y + w10 * v10.y + w11 * v11.y;
  o.z = w00 * v00.z + w01 * v01.z + w10 * v10.z + w11 * v11.z;
  o.w = w00 * v00.w + w01 * v01.w + w10 * v10.w + w11 * v11.w;
  ((float4*)dst)[gid] = o;
}

// ---------------------------------------------------------------------------
extern "C" void kernel_launch(void* const* d_in, const int* in_sizes, int n_in,
                              void* d_out, int out_size, void* d_ws, size_t ws_size,
                              hipStream_t stream) {
  (void)in_sizes; (void)n_in; (void)out_size; (void)ws_size;
  const float* f_t0    = (const float*)d_in[0];
  const float* f_t1    = (const float*)d_in[1];
  const float* enc_w1  = (const float*)d_in[2];
  const float* enc_b1  = (const float*)d_in[3];
  const float* ln1_g   = (const float*)d_in[4];
  const float* ln1_b   = (const float*)d_in[5];
  const float* enc_w2  = (const float*)d_in[6];
  const float* enc_b2  = (const float*)d_in[7];
  const float* enc_w3  = (const float*)d_in[8];
  const float* enc_b3  = (const float*)d_in[9];
  const float* ln2_g   = (const float*)d_in[10];
  const float* ln2_b   = (const float*)d_in[11];
  const float* emb     = (const float*)d_in[12];
  const float* dec_w1  = (const float*)d_in[13];
  const float* dec_b1  = (const float*)d_in[14];
  const float* dec_w2  = (const float*)d_in[15];
  const float* dec_b2  = (const float*)d_in[16];

  char* ws = (char*)d_ws;
  bf16_t* delta = (bf16_t*)(ws + OFF_DELTA);
  bf16_t* x1    = (bf16_t*)(ws + OFF_X1);
  float*  y1    = (float*)(ws + OFF_Y1);
  bf16_t* x2    = (bf16_t*)(ws + OFF_X2);
  float*  y3    = (float*)(ws + OFF_Y3);
  float*  x3    = (float*)(ws + OFF_X3);
  float*  poolf = (float*)(ws + OFF_POOLF);
  bf16_t* poolb = (bf16_t*)(ws + OFF_POOLB);
  bf16_t* qb    = (bf16_t*)(ws + OFF_QB);
  bf16_t* h1    = (bf16_t*)(ws + OFF_H1);
  float*  ydec  = (float*)(ws + OFF_YDEC);
  bf16_t* w1t   = (bf16_t*)(ws + OFF_W1T);
  bf16_t* w2t   = (bf16_t*)(ws + OFF_W2T);
  bf16_t* w3t   = (bf16_t*)(ws + OFF_W3T);
  bf16_t* dw1t  = (bf16_t*)(ws + OFF_DW1T);
  bf16_t* dw2t  = (bf16_t*)(ws + OFF_DW2T);
  bf16_t* embb  = (bf16_t*)(ws + OFF_EMBB);
  float*  ehalf = (float*)(ws + OFF_EHALF);
  float*  avg   = (float*)(ws + OFF_AVG);
  float*  cmt   = (float*)(ws + OFF_CMT);
  int*    idxi  = (int*)(ws + OFF_IDX);
  float*  outf  = (float*)d_out;

  // ---- prep: delta, weight transposes/conversions, codebook norms ----
  delta_bf16_k<<<(NTOK * 512 + 255) / 256, 256, 0, stream>>>(f_t0, f_t1, delta,
                                                             (size_t)NTOK * 512);
  transp_bf16_k<<<(512 * 512 + 255) / 256, 256, 0, stream>>>(enc_w1, w1t, 512, 512);
  transp_bf16_k<<<(512 * 128 + 255) / 256, 256, 0, stream>>>(enc_w2, w2t, 512, 128);
  transp_bf16_k<<<(128 * 64 + 255) / 256, 256, 0, stream>>>(enc_w3, w3t, 128, 64);
  transp_bf16_k<<<(64 * 128 + 255) / 256, 256, 0, stream>>>(dec_w1, dw1t, 64, 128);
  transp_bf16_k<<<(128 * 512 + 255) / 256, 256, 0, stream>>>(dec_w2, dw2t, 128, 512);
  emb_prep_k<<<NE, 32, 0, stream>>>(emb, embb, ehalf);
  zero_k<<<(NE + 64 + 255) / 256, 256, 0, stream>>>(avg, NE + 16);  // avg + commit

  // ---- encoder ----
  gemm_wmma<4, true, false, false><<<dim3(NTOK / 16, 512 / 64), 32, 0, stream>>>(
      delta, w1t, enc_b1, y1, nullptr, 512, 512);
  ln1_gelu_k<<<NTOK, 128, 0, stream>>>(y1, ln1_g, ln1_b, x1);
  gemm_wmma<4, true, true, true><<<dim3(NTOK / 16, 128 / 64), 32, 0, stream>>>(
      x1, w2t, enc_b2, nullptr, x2, 128, 512);
  gemm_wmma<4, true, false, false><<<dim3(NTOK / 16, 1), 32, 0, stream>>>(
      x2, w3t, enc_b3, y3, nullptr, 64, 128);
  ln2_k<<<NTOK, 32, 0, stream>>>(y3, ln2_g, ln2_b, x3);
  pool_norm_k<<<NQ, 32, 0, stream>>>(x3, poolf, poolb);

  // ---- VQ ----
  vq_k<<<NQ / 16, 32, 0, stream>>>(poolb, embb, ehalf, poolf, emb, avg, cmt,
                                   idxi, outf + OUT_DEC + 2, qb);
  vq_finalize_k<<<1, 256, 0, stream>>>(avg, cmt, outf);

  // ---- decoder ----
  gemm_wmma<4, true, true, true><<<dim3(NQ / 16, 128 / 64), 32, 0, stream>>>(
      qb, dw1t, dec_b1, nullptr, h1, 128, 64);
  gemm_wmma<4, true, false, false><<<dim3(NQ / 16, 512 / 64), 32, 0, stream>>>(
      h1, dw2t, dec_b2, ydec, nullptr, 512, 128);

  // ---- bilinear 2x upsample into d_out ----
  upsample2x_k<<<(unsigned)((size_t)32 * 32 * 32 * 128 / 256), 256, 0, stream>>>(
      ydec, outf);
}
